// AgentCrossAttention_6141803233369
// MI455X (gfx1250) — matrix operand order repaired and checked by reading, not compiled, not run on hardware
//
#include <hip/hip_runtime.h>
#include <hip/hip_bf16.h>
#include <math.h>

#define DIM 1024
#define NHEADS 16
#define NKV 4
#define HD 64
#define SS 256
#define KVD 256
#define SOFTCAP 50.0f
#define EPS 1e-6f

#define ASTR 40   // Abuf row stride (bf16 units); 80 B = 16B-aligned rows
#define BSTR 40   // BkT/BvT row stride; 80 B keeps fragment reads conflict-light
#define SLAB 128  // N-slab width: 2 passes over z instead of 4

typedef __bf16 bf16_t;
typedef bf16_t v16bf __attribute__((ext_vector_type(16)));
typedef float  v8f   __attribute__((ext_vector_type(8)));

union FragBF { unsigned u[8]; uint4 q[2]; v16bf v; };

__device__ __forceinline__ unsigned short f2bf(float f) {
  unsigned u = __builtin_bit_cast(unsigned, f);
  u += 0x7FFFu + ((u >> 16) & 1u);           // round-to-nearest-even
  return (unsigned short)(u >> 16);
}
__device__ __forceinline__ float bf2f(unsigned short h) {
  unsigned u = ((unsigned)h) << 16;
  return __builtin_bit_cast(float, u);
}

// ---------------------------------------------------------------------------
// Kernel 0: one-time  W[1024][256] fp32  ->  WT[256][1024] bf16  (tiled)
// ---------------------------------------------------------------------------
__global__ __launch_bounds__(256)
void wtranspose_bf16_kernel(const float* __restrict__ W,
                            unsigned short* __restrict__ WT) {
  __shared__ float tile[64][65];
  const int tx = threadIdx.x & 63, ty = threadIdx.x >> 6;   // 64 x 4
  const int k0 = (blockIdx.x & 15) * 64;                    // 16 k-tiles
  const int n0 = (blockIdx.x >> 4) * 64;                    // 4  n-tiles
  for (int i = 0; i < 16; ++i) {
    int k = ty + i * 4;
    tile[k][tx] = W[(size_t)(k0 + k) * KVD + n0 + tx];
  }
  __syncthreads();
  for (int i = 0; i < 16; ++i) {
    int n = ty + i * 4;
    WT[(size_t)(n0 + n) * DIM + k0 + tx] = f2bf(tile[tx][n]);
  }
}

// ---------------------------------------------------------------------------
// Kernel 1: Q = rmsnorm_per_head(agent @ Wq) * q_gamma   (fp32, 1.07 GFLOP)
// ---------------------------------------------------------------------------
__global__ __launch_bounds__(256)
void qproj_rmsnorm_kernel(const float* __restrict__ agent,
                          const float* __restrict__ Wq,
                          const float* __restrict__ qg,
                          float* __restrict__ qout) {
  __shared__ float arow[8][DIM];
  __shared__ float red[8][NHEADS];
  __shared__ float rsig[8][NHEADS];
  const int tid = threadIdx.x;
  const int r0 = blockIdx.x * 8;
  for (int i = 0; i < 32; ++i) {
    int idx = tid + i * 256;
    int r = idx >> 10, c = idx & 1023;
    arow[r][c] = agent[(size_t)(r0 + r) * DIM + c];
  }
  if (tid < 8 * NHEADS) red[tid >> 4][tid & 15] = 0.f;
  __syncthreads();

  float acc[8][4];
  for (int r = 0; r < 8; ++r) for (int j = 0; j < 4; ++j) acc[r][j] = 0.f;
  for (int d = 0; d < DIM; ++d) {
    const float* wr = Wq + (size_t)d * DIM;
    float w0 = wr[tid], w1 = wr[tid + 256], w2 = wr[tid + 512], w3 = wr[tid + 768];
    for (int r = 0; r < 8; ++r) {
      float a = arow[r][d];
      acc[r][0] = fmaf(a, w0, acc[r][0]);
      acc[r][1] = fmaf(a, w1, acc[r][1]);
      acc[r][2] = fmaf(a, w2, acc[r][2]);
      acc[r][3] = fmaf(a, w3, acc[r][3]);
    }
  }
  for (int r = 0; r < 8; ++r)
    for (int j = 0; j < 4; ++j) {
      int c = tid + j * 256;
      atomicAdd(&red[r][c >> 6], acc[r][j] * acc[r][j]);
    }
  __syncthreads();
  if (tid < 8 * NHEADS) {
    int r = tid >> 4, h = tid & 15;
    rsig[r][h] = rsqrtf(red[r][h] * (1.0f / HD) + EPS);
  }
  __syncthreads();
  for (int r = 0; r < 8; ++r)
    for (int j = 0; j < 4; ++j) {
      int c = tid + j * 256;
      qout[(size_t)(r0 + r) * DIM + c] = acc[r][j] * rsig[r][c >> 6] * qg[c & 63];
    }
}

// ---------------------------------------------------------------------------
// Kernel 2: per-(b,t) fused  K/V projection (bf16 WMMA) + K-rmsnorm +
// softcapped GQA attention.  One 256-thread (8-wave) block per (b,t).
// 2 N-passes of 128; 32 WMMA per staged chunk per wave.  LDS = 322,752 B.
// ---------------------------------------------------------------------------
__global__ __launch_bounds__(256)
void kv_attn_kernel(const float* __restrict__ z,
                    const unsigned short* __restrict__ WkT,  // [256][1024] bf16
                    const unsigned short* __restrict__ WvT,  // [256][1024] bf16
                    const float* __restrict__ kg,
                    const float* __restrict__ qbuf,
                    float* __restrict__ attn_out) {
  extern __shared__ char smem[];
  unsigned short* Kbf  = (unsigned short*)smem;       // [256][258] bf16
  unsigned short* Vbf  = Kbf + 256 * 258;             // [256][258]
  unsigned short* Abuf = Vbf + 256 * 258;             // [256][ASTR] z chunk bf16
  unsigned short* BkT  = Abuf + 256 * ASTR;           // [SLAB][BSTR] Wk slab (n-major)
  unsigned short* BvT  = BkT + SLAB * BSTR;           // [SLAB][BSTR]
  float* scores = (float*)(BvT + SLAB * BSTR);        // [16][256]
  float* red    = scores + 16 * 256;                  // [16][17]
  float* rowmax = red + 16 * 17;                      // [16]
  float* rowsum = rowmax + 16;                        // [16]
  float* qrow   = (float*)Abuf;                       // reuse after GEMM (4 KB)

  const int tid = threadIdx.x;
  const int lane = tid & 31, wid = tid >> 5;
  const int lhalf = lane >> 4, l16 = lane & 15;
  const int wm = wid & 3, wn = wid >> 2;              // 4(M) x 2(N) wave grid
  const int bt = blockIdx.x;
  const float* zt = z + (size_t)bt * SS * DIM;

  // ---- K/V projection: (256 x 1024) @ (1024 x 256), bf16 WMMA ----
  for (int slab = 0; slab < 2; ++slab) {              // 128-wide N slabs
    v8f accK[4][4], accV[4][4];
    for (int mt = 0; mt < 4; ++mt)
      for (int nt = 0; nt < 4; ++nt) { accK[mt][nt] = v8f{}; accV[mt][nt] = v8f{}; }

    // B staging indices: 2 threads per n-row, 2 x uint4 each
    const int bn = tid >> 1;                          // 0..127
    const int bj = (tid & 1) * 16;                    // 0 or 16 (u16 units)
    const size_t bgoff = (size_t)(slab * SLAB + bn) * DIM + bj;

    for (int kk = 0; kk < DIM; kk += 32) {
      { // stage z chunk (256x32 fp32 -> bf16, row-major, 8B stores)
        int cr = tid >> 3;                            // 0..31
        int c4 = (tid & 7) * 4;                       // 0..28
        for (int rr = 0; rr < 8; ++rr) {
          int row = rr * 32 + cr;
          const float4 f = *(const float4*)(zt + (size_t)row * DIM + kk + c4);
          uint2 p;
          p.x = (unsigned)f2bf(f.x) | ((unsigned)f2bf(f.y) << 16);
          p.y = (unsigned)f2bf(f.z) | ((unsigned)f2bf(f.w) << 16);
          *(uint2*)&Abuf[row * ASTR + c4] = p;
          if (kk + 32 < DIM)
            __builtin_prefetch(zt + (size_t)row * DIM + kk + 32 + c4, 0, 0);
        }
      }
      { // stage Wk/Wv chunk: pure wide copy of pre-converted bf16
        *(uint4*)&BkT[bn * BSTR + bj]     = *(const uint4*)&WkT[bgoff + kk];
        *(uint4*)&BkT[bn * BSTR + bj + 8] = *(const uint4*)&WkT[bgoff + kk + 8];
        *(uint4*)&BvT[bn * BSTR + bj]     = *(const uint4*)&WvT[bgoff + kk];
        *(uint4*)&BvT[bn * BSTR + bj + 8] = *(const uint4*)&WvT[bgoff + kk + 8];
      }
      __syncthreads();

      FragBF a[4], bk[4], bv[4];
      for (int mt = 0; mt < 4; ++mt) {                // A: 16x32 bf16 layout
        int row = wm * 64 + mt * 16 + l16;
        a[mt].q[0] = *(const uint4*)&Abuf[row * ASTR + lhalf * 16];
        a[mt].q[1] = *(const uint4*)&Abuf[row * ASTR + lhalf * 16 + 8];
      }
      for (int nt = 0; nt < 4; ++nt) {                // B: 32x16 bf16 layout
        int n = wn * 64 + nt * 16 + l16;
        bk[nt].q[0] = *(const uint4*)&BkT[n * BSTR + lhalf * 16];
        bk[nt].q[1] = *(const uint4*)&BkT[n * BSTR + lhalf * 16 + 8];
        bv[nt].q[0] = *(const uint4*)&BvT[n * BSTR + lhalf * 16];
        bv[nt].q[1] = *(const uint4*)&BvT[n * BSTR + lhalf * 16 + 8];
      }
      for (int mt = 0; mt < 4; ++mt)
        for (int nt = 0; nt < 4; ++nt) {
          accK[mt][nt] = __builtin_amdgcn_wmma_f32_16x16x32_bf16(
              false, a[mt].v, false, bk[nt].v, (short)0, accK[mt][nt], false, false);
          accV[mt][nt] = __builtin_amdgcn_wmma_f32_16x16x32_bf16(
              false, a[mt].v, false, bv[nt].v, (short)0, accV[mt][nt], false, false);
        }
      __syncthreads();
    }
    // spill slab accumulators to LDS as bf16 (C layout: VGPR r -> M=r+half*8)
    for (int mt = 0; mt < 4; ++mt)
      for (int nt = 0; nt < 4; ++nt)
        for (int r = 0; r < 8; ++r) {
          int srow = wm * 64 + mt * 16 + r + lhalf * 8;
          int c = slab * SLAB + wn * 64 + nt * 16 + l16;
          Kbf[srow * 258 + c] = f2bf(accK[mt][nt][r]);
          Vbf[srow * 258 + c] = f2bf(accV[mt][nt][r]);
        }
    __syncthreads();
  }

  // ---- RMSNorm K per (s, kv-head) ----
  for (int i = 0; i < 4; ++i) {
    int task = tid + i * 256;
    int s = task & 255, kvh = task >> 8;
    unsigned short* kp = &Kbf[s * 258 + kvh * 64];
    float sq = 0.f;
    for (int d = 0; d < HD; ++d) { float v = bf2f(kp[d]); sq = fmaf(v, v, sq); }
    float rs = rsqrtf(sq * (1.0f / HD) + EPS);
    for (int d = 0; d < HD; ++d) kp[d] = f2bf(bf2f(kp[d]) * rs * kg[d]);
  }
  __syncthreads();

  // ---- load normalized q row (overlaps Abuf: GEMM done) ----
  for (int i = 0; i < 4; ++i) {
    int c = tid + i * 256;
    qrow[c] = qbuf[(size_t)bt * DIM + c];
  }
  __syncthreads();

  // ---- scores: 16 heads x 256 keys, softcap ----
  const int h = tid >> 4, si = tid & 15;
  const int kvh = h >> 2;                              // GQA: G=4
  for (int i = 0; i < 16; ++i) {
    int s = si * 16 + i;
    const unsigned short* kp = &Kbf[s * 258 + kvh * 64];
    const float* qp = &qrow[h * 64];
    float acc = 0.f;
    for (int d = 0; d < HD; ++d) acc = fmaf(qp[d], bf2f(kp[d]), acc);
    float sc = acc * 0.125f;                           // hd^-0.5
    scores[h * 256 + s] = SOFTCAP * tanhf(sc * (1.0f / SOFTCAP));
  }
  __syncthreads();

  // ---- softmax over s (16 threads per head) ----
  float lmax = -1e30f;
  for (int i = 0; i < 16; ++i) lmax = fmaxf(lmax, scores[h * 256 + si * 16 + i]);
  red[h * 17 + si] = lmax;
  __syncthreads();
  if (si == 0) {
    float m = red[h * 17];
    for (int j = 1; j < 16; ++j) m = fmaxf(m, red[h * 17 + j]);
    rowmax[h] = m;
  }
  __syncthreads();
  float m = rowmax[h], lsum = 0.f;
  for (int i = 0; i < 16; ++i) {
    int idx = h * 256 + si * 16 + i;
    float e = __expf(scores[idx] - m);
    scores[idx] = e;
    lsum += e;
  }
  red[h * 17 + si] = lsum;
  __syncthreads();
  if (si == 0) {
    float sum = 0.f;
    for (int j = 0; j < 16; ++j) sum += red[h * 17 + j];
    rowsum[h] = sum;
  }
  __syncthreads();
  float inv = 1.0f / rowsum[h];
  for (int i = 0; i < 16; ++i) scores[h * 256 + si * 16 + i] *= inv;
  __syncthreads();

  // ---- out = attn @ V ----
  for (int j = 0; j < 4; ++j) {
    int d = si * 4 + j;
    float acc = 0.f;
    for (int s = 0; s < SS; ++s)
      acc = fmaf(scores[h * 256 + s], bf2f(Vbf[s * 258 + kvh * 64 + d]), acc);
    attn_out[(size_t)bt * DIM + h * 64 + d] = acc;
  }
}

// ---------------------------------------------------------------------------
// Kernel 3: out = attn_out @ Wo   (fp32, 1.07 GFLOP), 8 rows per block
// ---------------------------------------------------------------------------
__global__ __launch_bounds__(256)
void oproj_kernel(const float* __restrict__ xin,
                  const float* __restrict__ Wo,
                  float* __restrict__ out) {
  __shared__ float arow[8][DIM];
  const int tid = threadIdx.x;
  const int r0 = blockIdx.x * 8;
  for (int i = 0; i < 32; ++i) {
    int idx = tid + i * 256;
    int r = idx >> 10, c = idx & 1023;
    arow[r][c] = xin[(size_t)(r0 + r) * DIM + c];
  }
  __syncthreads();
  float acc[8][4];
  for (int r = 0; r < 8; ++r) for (int j = 0; j < 4; ++j) acc[r][j] = 0.f;
  for (int d = 0; d < DIM; ++d) {
    const float* wr = Wo + (size_t)d * DIM;
    float w0 = wr[tid], w1 = wr[tid + 256], w2 = wr[tid + 512], w3 = wr[tid + 768];
    for (int r = 0; r < 8; ++r) {
      float a = arow[r][d];
      acc[r][0] = fmaf(a, w0, acc[r][0]);
      acc[r][1] = fmaf(a, w1, acc[r][1]);
      acc[r][2] = fmaf(a, w2, acc[r][2]);
      acc[r][3] = fmaf(a, w3, acc[r][3]);
    }
  }
  for (int r = 0; r < 8; ++r)
    for (int j = 0; j < 4; ++j)
      out[(size_t)(r0 + r) * DIM + tid + j * 256] = acc[r][j];
}

// ---------------------------------------------------------------------------
extern "C" void kernel_launch(void* const* d_in, const int* in_sizes, int n_in,
                              void* d_out, int out_size, void* d_ws, size_t ws_size,
                              hipStream_t stream) {
  const float* agent = (const float*)d_in[0];
  const float* z     = (const float*)d_in[1];
  const float* Wq    = (const float*)d_in[2];
  const float* Wk    = (const float*)d_in[3];
  const float* Wv    = (const float*)d_in[4];
  const float* Wo    = (const float*)d_in[5];
  const float* qg    = (const float*)d_in[6];
  const float* kg    = (const float*)d_in[7];
  const int BT = in_sizes[0] / DIM;                    // 512

  float* q_ws    = (float*)d_ws;                       // BT*1024 f32 (2 MB)
  float* attn_ws = q_ws + (size_t)BT * DIM;            // BT*1024 f32 (2 MB)
  unsigned short* WkT = (unsigned short*)(attn_ws + (size_t)BT * DIM); // 512 KB
  unsigned short* WvT = WkT + (size_t)KVD * DIM;                       // 512 KB

  wtranspose_bf16_kernel<<<64, 256, 0, stream>>>(Wk, WkT);
  wtranspose_bf16_kernel<<<64, 256, 0, stream>>>(Wv, WvT);
  qproj_rmsnorm_kernel<<<BT / 8, 256, 0, stream>>>(agent, Wq, qg, q_ws);

  const size_t smem2 = (size_t)(2 * 256 * 258 + 256 * ASTR + 2 * SLAB * BSTR) * 2 +
                       (size_t)(16 * 256 + 16 * 17 + 32) * 4;   // 322,752 B
  kv_attn_kernel<<<BT, 256, smem2, stream>>>(z, WkT, WvT, kg, q_ws, attn_ws);

  oproj_kernel<<<BT / 8, 256, 0, stream>>>(attn_ws, Wo, (float*)d_out);
}
